// AttentionBlock_13529146983011
// MI455X (gfx1250) — matrix operand order repaired and checked by reading, not compiled, not run on hardware
//
#include <hip/hip_runtime.h>

#define BB 4
#define TT 4096
#define CC 128
#define KK 64
#define VV 64
#define EPS 1e-5f

typedef __attribute__((ext_vector_type(16))) __bf16          v16bf;
typedef __attribute__((ext_vector_type(16))) unsigned short  v16us;
typedef __attribute__((ext_vector_type(8)))  float           v8f;
typedef __attribute__((ext_vector_type(4)))  unsigned int    v4u;
typedef __attribute__((ext_vector_type(8)))  unsigned int    v8u;

#define VT_STRIDE 72   // ushorts; 144B rows: 16B-aligned, not a multiple of 256B (banks)

// ---------- helpers ----------
__device__ __forceinline__ unsigned short f32_to_bf16(float f) {
    unsigned int u = __builtin_bit_cast(unsigned int, f);
    u += 0x7FFFu + ((u >> 16) & 1u);   // round-to-nearest-even
    return (unsigned short)(u >> 16);
}

// wave32 xor-swizzle (group-of-32 mode: offset[14:10]=xor, [4:0]=and)
template <int XMASK>
__device__ __forceinline__ float swz(float x) {
    int r = __builtin_amdgcn_ds_swizzle(__builtin_bit_cast(int, x),
                                        (XMASK << 10) | 0x1F);
    return __builtin_bit_cast(float, r);
}
__device__ __forceinline__ float rmax16(float v) {           // reduce within 16-lane half
    v = fmaxf(v, swz<1>(v)); v = fmaxf(v, swz<2>(v));
    v = fmaxf(v, swz<4>(v)); v = fmaxf(v, swz<8>(v));
    return v;
}
__device__ __forceinline__ float rsum16(float v) {
    v += swz<1>(v); v += swz<2>(v); v += swz<4>(v); v += swz<8>(v);
    return v;
}
__device__ __forceinline__ float rsum32(float v) {
    float t = rsum16(v);
    return t + swz<16>(t);
}

// ---- Tensor Data Mover: async 2D tile load (64 rows x 64 bf16) global -> LDS ----
// D# per CDNA5 ISA:
//   group0: [0]=count=1, [1]=lds byte addr, [2]=global_addr[31:0],
//           [3]=global_addr[56:32] | type(2)<<30
//   group1: [0]=data_size(2B)<<16, [1]=tensor_dim0(64)<<16,
//           [2]=tensor_dim1(T)<<16, [3]=tile_dim0(64)<<16,
//           [4]=tile_dim1(64), [5]=tensor_dim0_stride(64), [6..7]=0
__device__ __forceinline__ void tdm_load_k_tile(const unsigned short* gsrc,
                                                unsigned int lds_addr) {
    const unsigned long long ga = (unsigned long long)(uintptr_t)gsrc;
    v4u g0;
    g0[0] = 1u;
    g0[1] = lds_addr;
    g0[2] = (unsigned int)ga;
    g0[3] = (unsigned int)(ga >> 32) | (2u << 30);
    v8u g1;
    g1[0] = 1u << 16;                  // data_size = 2 bytes
    g1[1] = 64u << 16;                 // tensor_dim0 = 64 elements
    g1[2] = (unsigned int)TT << 16;    // tensor_dim1 = T rows
    g1[3] = 64u << 16;                 // tile_dim0 = 64
    g1[4] = 64u;                       // tile_dim1 = 64, tile_dim2 = 0
    g1[5] = 64u;                       // tensor_dim0_stride = 64 elements
    g1[6] = 0u;
    g1[7] = 0u;
    asm volatile("tensor_load_to_lds %0, %1" :: "s"(g0), "s"(g1) : "memory");
}

// ---------- kernel 1: projections + LN -> bf16 q,k,v ; copy x into out[:, :128] ----------
__global__ __launch_bounds__(256)
void proj_ln_kernel(const float* __restrict__ x,
                    const float* __restrict__ Wq, const float* __restrict__ bq,
                    const float* __restrict__ Wk, const float* __restrict__ bk,
                    const float* __restrict__ Wv, const float* __restrict__ bv,
                    const float* __restrict__ gq, const float* __restrict__ bgq,
                    const float* __restrict__ gk, const float* __restrict__ bgk,
                    const float* __restrict__ gv, const float* __restrict__ bgv,
                    unsigned short* __restrict__ qo,
                    unsigned short* __restrict__ ko,
                    unsigned short* __restrict__ vo,
                    float* __restrict__ out)
{
    __shared__ float xs[8][CC];
    const int wave = threadIdx.x >> 5;
    const int lane = threadIdx.x & 31;
    const long token = (long)blockIdx.x * 8 + wave;       // b*T + t

    const float* xr = x + token * CC;
    float* orow = out + token * (CC + VV);
    #pragma unroll
    for (int c = lane; c < CC; c += 32) {
        float xv = xr[c];
        xs[wave][c] = xv;
        orow[c] = xv;                                     // concat: out[:, :128] = x
    }
    __syncthreads();

    const int j0 = lane, j1 = lane + 32;                  // each lane owns 2 output dims
    float aq0 = bq[j0], aq1 = bq[j1];
    float ak0 = bk[j0], ak1 = bk[j1];
    float av0 = bv[j0], av1 = bv[j1];
    #pragma unroll 4
    for (int c = 0; c < CC; ++c) {
        const float xv = xs[wave][c];
        aq0 = fmaf(xv, Wq[c * KK + j0], aq0);
        aq1 = fmaf(xv, Wq[c * KK + j1], aq1);
        ak0 = fmaf(xv, Wk[c * KK + j0], ak0);
        ak1 = fmaf(xv, Wk[c * KK + j1], ak1);
        av0 = fmaf(xv, Wv[c * VV + j0], av0);
        av1 = fmaf(xv, Wv[c * VV + j1], av1);
    }

    // LayerNorm over 64 dims (spread 2/lane across 32 lanes)
    #define DO_LN(a0, a1, gg, bb, dst)                                          \
    {                                                                           \
        float mean = rsum32(a0 + a1) * (1.0f / 64.0f);                          \
        float var  = rsum32(a0 * a0 + a1 * a1) * (1.0f / 64.0f) - mean * mean;  \
        float rstd = rsqrtf(var + EPS);                                         \
        float y0 = (a0 - mean) * rstd * gg[j0] + bb[j0];                        \
        float y1 = (a1 - mean) * rstd * gg[j1] + bb[j1];                        \
        dst[token * 64 + j0] = f32_to_bf16(y0);                                 \
        dst[token * 64 + j1] = f32_to_bf16(y1);                                 \
    }
    DO_LN(aq0, aq1, gq, bgq, qo)
    DO_LN(ak0, ak1, gk, bgk, ko)
    DO_LN(av0, av1, gv, bgv, vo)
    #undef DO_LN
}

// ---------- kernel 2: causal flash attention (bf16 WMMA, TDM-staged K) + output LN ----------
__global__ __launch_bounds__(128)
void attn_kernel(const unsigned short* __restrict__ q,
                 const unsigned short* __restrict__ k,
                 const unsigned short* __restrict__ v,
                 const float* __restrict__ go, const float* __restrict__ bgo,
                 float* __restrict__ out)
{
    __shared__ unsigned short Ks[64 * 64];           // key tile, row-major (TDM target)
    __shared__ unsigned short Vt[64 * VT_STRIDE];    // value tile, TRANSPOSED: [feat][key]
    __shared__ unsigned short Ps[4][16 * 64];        // per-wave P scratch

    const int b    = blockIdx.y;
    const int r0   = blockIdx.x * 64;                // workgroup's first query row
    const int wave = threadIdx.x >> 5;
    const int lane = threadIdx.x & 31;
    const int g    = lane >> 4;                      // lane half
    const int ln   = lane & 15;
    const int rbase = r0 + wave * 16;                // this wave's first query row

    // --- load Q as A-fragments (16 rows x 64 feat, 2 K-steps of 32) ---
    const unsigned short* qb = q + ((long)b * TT + rbase) * 64;
    v16us aq[2];
    #pragma unroll
    for (int ks = 0; ks < 2; ++ks)
        #pragma unroll
        for (int j = 0; j < 8; ++j) {
            aq[ks][j]     = qb[(long)ln * 64 + ks * 32 + 16 * g + j];
            aq[ks][j + 8] = qb[(long)ln * 64 + ks * 32 + 16 * g + 8 + j];
        }

    v8f o[4] = {v8f{0}, v8f{0}, v8f{0}, v8f{0}};
    float m_i[8], l_i[8];
    #pragma unroll
    for (int i = 0; i < 8; ++i) { m_i[i] = -1e30f; l_i[i] = 0.0f; }

    const unsigned int* v32 = (const unsigned int*)v;
    unsigned short* pw = Ps[wave];
    const unsigned int ksAddr = (unsigned int)(uintptr_t)(void*)&Ks[0]; // LDS aperture: low 32 bits = LDS offset

    const int nTiles = blockIdx.x + 1;               // causal: keys up to r0+63
    for (int t0 = 0; t0 < nTiles; ++t0) {
        const int s0 = t0 * 64;

        // K tile: Tensor Data Mover, issued once by wave 0 (EXEC-independent DMA)
        if (wave == 0)
            tdm_load_k_tile(k + ((long)b * TT + s0) * 64, ksAddr);

        // V tile: cooperative transposed load (coalesced global dwords -> LDS scatter)
        for (int i = threadIdx.x; i < 2048; i += 128) {
            const int row = i >> 5;                  // key index 0..63
            const int fp  = i & 31;                  // feature pair
            const unsigned int w = v32[((long)b * TT + s0 + row) * 32 + fp];
            Vt[(2 * fp)     * VT_STRIDE + row] = (unsigned short)w;
            Vt[(2 * fp + 1) * VT_STRIDE + row] = (unsigned short)(w >> 16);
        }
        if (wave == 0)
            __builtin_amdgcn_s_wait_tensorcnt(0);    // K tile DMA complete
        __syncthreads();

        // ---- S = Q * K^T  (4 N-tiles x 2 K-steps) ----
        v8f s[4] = {v8f{0}, v8f{0}, v8f{0}, v8f{0}};
        #pragma unroll
        for (int ks = 0; ks < 2; ++ks) {
            const v16bf A = __builtin_bit_cast(v16bf, aq[ks]);
            #pragma unroll
            for (int nt = 0; nt < 4; ++nt) {
                v16us bfrag;
                const unsigned short* kr = &Ks[(nt * 16 + ln) * 64 + ks * 32 + 16 * g];
                #pragma unroll
                for (int j = 0; j < 16; ++j) bfrag[j] = kr[j];
                s[nt] = __builtin_amdgcn_wmma_f32_16x16x32_bf16(
                            false, A, false, __builtin_bit_cast(v16bf, bfrag),
                            (short)0, s[nt], false, false);
            }
        }

        // ---- scale + causal mask (C-layout: row = rbase+i+8g, col = s0+nt*16+ln) ----
        #pragma unroll
        for (int nt = 0; nt < 4; ++nt)
            #pragma unroll
            for (int i = 0; i < 8; ++i) {
                float sv = s[nt][i] * 0.125f;        // 1/sqrt(64)
                const int row = rbase + i + 8 * g;
                const int col = s0 + nt * 16 + ln;
                s[nt][i] = (col > row) ? -1e30f : sv;
            }

        // ---- online softmax ----
        float mnew[8], scl[8];
        #pragma unroll
        for (int i = 0; i < 8; ++i) {
            float mx = fmaxf(fmaxf(s[0][i], s[1][i]), fmaxf(s[2][i], s[3][i]));
            mx = rmax16(mx);
            mnew[i] = fmaxf(m_i[i], mx);
            scl[i]  = __expf(m_i[i] - mnew[i]);
        }
        float rs[8];
        #pragma unroll
        for (int i = 0; i < 8; ++i) rs[i] = 0.0f;
        #pragma unroll
        for (int nt = 0; nt < 4; ++nt)
            #pragma unroll
            for (int i = 0; i < 8; ++i) {
                const float p = __expf(s[nt][i] - mnew[i]);
                s[nt][i] = p;
                rs[i] += p;
            }
        #pragma unroll
        for (int i = 0; i < 8; ++i) {
            l_i[i] = l_i[i] * scl[i] + rsum16(rs[i]);
            m_i[i] = mnew[i];
            #pragma unroll
            for (int nt = 0; nt < 4; ++nt) o[nt][i] *= scl[i];
        }

        // ---- P: C-layout -> LDS (row-major 16x64 bf16) ----
        #pragma unroll
        for (int nt = 0; nt < 4; ++nt)
            #pragma unroll
            for (int i = 0; i < 8; ++i)
                pw[(i + 8 * g) * 64 + nt * 16 + ln] = f32_to_bf16(s[nt][i]);
        __syncthreads();

        // ---- O += P * V : A from Ps (contiguous), B from transposed Vt (contiguous) ----
        #pragma unroll
        for (int ks = 0; ks < 2; ++ks) {
            v16us ap;
            #pragma unroll
            for (int j = 0; j < 8; ++j) {
                ap[j]     = pw[ln * 64 + ks * 32 + 16 * g + j];
                ap[j + 8] = pw[ln * 64 + ks * 32 + 16 * g + 8 + j];
            }
            const v16bf A = __builtin_bit_cast(v16bf, ap);
            #pragma unroll
            for (int nt = 0; nt < 4; ++nt) {
                v16us bfrag;
                const unsigned short* vr = &Vt[(nt * 16 + ln) * VT_STRIDE + ks * 32 + 16 * g];
                #pragma unroll
                for (int j = 0; j < 16; ++j) bfrag[j] = vr[j];
                o[nt] = __builtin_amdgcn_wmma_f32_16x16x32_bf16(
                            false, A, false, __builtin_bit_cast(v16bf, bfrag),
                            (short)0, o[nt], false, false);
            }
        }
        __syncthreads();   // K/Vt/P consumed; safe to overwrite next iteration
    }

    // ---- epilogue: O /= l, LayerNorm over 64 dims, write out[:, 128:192] ----
    #pragma unroll
    for (int i = 0; i < 8; ++i) {
        const float inv = 1.0f / l_i[i];
        float vals[4], sum = 0.0f, sq = 0.0f;
        #pragma unroll
        for (int nt = 0; nt < 4; ++nt) {
            const float xo = o[nt][i] * inv;
            vals[nt] = xo;
            sum += xo;
            sq  += xo * xo;
        }
        sum = rsum16(sum);
        sq  = rsum16(sq);
        const float mean = sum * (1.0f / 64.0f);
        const float var  = sq * (1.0f / 64.0f) - mean * mean;
        const float rstd = rsqrtf(var + EPS);
        const int row = rbase + i + 8 * g;
        float* orow = out + ((long)b * TT + row) * (CC + VV) + CC;
        #pragma unroll
        for (int nt = 0; nt < 4; ++nt) {
            const int n = nt * 16 + ln;
            orow[n] = (vals[nt] - mean) * rstd * go[n] + bgo[n];
        }
    }
}

// ---------- launcher ----------
extern "C" void kernel_launch(void* const* d_in, const int* in_sizes, int n_in,
                              void* d_out, int out_size, void* d_ws, size_t ws_size,
                              hipStream_t stream)
{
    const float* x   = (const float*)d_in[0];
    const float* Wq  = (const float*)d_in[1];
    const float* bq  = (const float*)d_in[2];
    const float* Wk  = (const float*)d_in[3];
    const float* bk  = (const float*)d_in[4];
    const float* Wv  = (const float*)d_in[5];
    const float* bv  = (const float*)d_in[6];
    const float* gq  = (const float*)d_in[7];
    const float* bgq = (const float*)d_in[8];
    const float* gk  = (const float*)d_in[9];
    const float* bgk = (const float*)d_in[10];
    const float* gv  = (const float*)d_in[11];
    const float* bgv = (const float*)d_in[12];
    const float* go  = (const float*)d_in[13];
    const float* bgo = (const float*)d_in[14];
    float* out = (float*)d_out;

    const size_t nqkv = (size_t)BB * TT * 64;             // elements per tensor
    unsigned short* qws = (unsigned short*)d_ws;
    unsigned short* kws = qws + nqkv;
    unsigned short* vws = kws + nqkv;

    // 1 wave per token, 8 tokens per block
    proj_ln_kernel<<<(BB * TT) / 8, 256, 0, stream>>>(
        x, Wq, bq, Wk, bk, Wv, bv, gq, bgq, gk, bgk, gv, bgv, qws, kws, vws, out);

    // 64 query rows per workgroup, 4 waves (16-row WMMA tiles each)
    attn_kernel<<<dim3(TT / 64, BB), 128, 0, stream>>>(
        qws, kws, vws, go, bgo, out);
}
